// VectorQuantizer_72859825209525
// MI455X (gfx1250) — matrix operand order repaired and checked by reading, not compiled, not run on hardware
//
#include <hip/hip_runtime.h>
#include <hip/hip_bf16.h>

// ---------------------------------------------------------------------------
// VQ argmin on gfx1250: split-bf16 WMMA GEMM (hi*hi + hi*lo + lo*hi, f32 acc)
// with TDM (tensor_load_to_lds) double-buffered codebook staging.
// ---------------------------------------------------------------------------

typedef __attribute__((ext_vector_type(16))) __bf16 v16bf;
typedef __attribute__((ext_vector_type(8)))  float  v8f;
typedef __attribute__((ext_vector_type(4)))  float  f4;
typedef __attribute__((ext_vector_type(4)))  unsigned int u32x4;
typedef __attribute__((ext_vector_type(8)))  int    i32x8;
typedef __attribute__((ext_vector_type(4)))  int    i32x4;

#define KCODES   8192
#define NROWS    16384      // B*H*W = 1*128*128
#define DDIM     64
#define CHUNK    128        // codes staged per TDM transfer
#define BPAD     144        // padded LDS stride per code (bytes): 36 dwords, conflict-free for 16-lane access
#define TDM_TILE 2048u      // CHUNK*128 bytes in 8-byte units
#define STALE_VALF 9990.0f

union Frag { v16bf v; f4 f[2]; };

__device__ __forceinline__ v8f wmma_bf16(v16bf a, v16bf b, v8f c) {
  return __builtin_amdgcn_wmma_f32_16x16x32_bf16(
      false, a, false, b, (short)0, c, false, false);
}

// ---------------------------------------------------------------------------
// TDM: DMA CHUNK*128 contiguous bytes from global into LDS, inserting a 16B
// pad after every 128B (-> BPAD=144 stride). D# per CDNA5 ISA:
//   group0: count=1 | lds_addr | global_addr[56:0] | type=2
//   group1: data_size=8B, pad_enable, pad_interval=2^4*8B=128B, pad_amount=4dw,
//           1-D tile: tile_dim0=2048, tensor_dim0=2048, tensor_dim1=1
//   remaining groups: zero (tensor <= 2D)
// This toolchain exposes the 6-arg builtin:
//   (u32x4 g0, i32x8 g1, i32x4, i32x4, i32x8, i32 cpol)
// ---------------------------------------------------------------------------
__device__ __forceinline__ void tdm_load_chunk(const void* gsrc, unsigned ldsOff) {
  const unsigned long long ga = (unsigned long long)(size_t)gsrc;
  u32x4 g0;
  g0[0] = 1u;                                                 // count=1, user desc
  g0[1] = ldsOff;                                             // lds_addr (bytes)
  g0[2] = (unsigned)ga;                                       // global_addr[31:0]
  g0[3] = (unsigned)((ga >> 32) & 0x01FFFFFFu) | (2u << 30);  // addr[56:32] | type=2

  i32x8 g1;
  g1[0] = (int)((3u << 16) | (1u << 20) | (4u << 22) | (3u << 25)); // dsize=8B|pad_en|intv|amt
  g1[1] = (int)((TDM_TILE & 0xFFFFu) << 16);                  // tensor_dim0[15:0] @ bits 63:48
  g1[2] = (int)((TDM_TILE >> 16) | (1u << 16));               // tensor_dim0[31:16] | tensor_dim1=1
  g1[3] = (int)(TDM_TILE << 16);                              // tile_dim0 @ bits 127:112
  g1[4] = 0;                                                  // tile_dim1/2 unused
  g1[5] = (int)TDM_TILE;                                      // tensor_dim0_stride lo32
  g1[6] = 0;
  g1[7] = 0;

  i32x4 z4 = {0, 0, 0, 0};
  i32x8 z8 = {0, 0, 0, 0, 0, 0, 0, 0};
  __builtin_amdgcn_tensor_load_to_lds(g0, g1, z4, z4, z8, 0);
}

// ---------------------------------------------------------------------------
__global__ __launch_bounds__(256) void prep_codes(
    const float* __restrict__ embed,   // [64][8193]
    const int*   __restrict__ cnt,     // [8192]
    __bf16* __restrict__ chi,          // [8192][64]
    __bf16* __restrict__ clo,          // [8192][64]
    float*  __restrict__ cf32,         // [64][8192]
    float*  __restrict__ cnorm)        // [8192]
{
  const int k = blockIdx.x * 256 + threadIdx.x;
  if (k >= KCODES) return;
  const bool stale = cnt[k] < 1;
  float norm = 0.f;
#pragma unroll
  for (int d = 0; d < DDIM; ++d) {
    float v = stale ? STALE_VALF : embed[(size_t)d * (KCODES + 1) + k];
    norm += v * v;
    __bf16 hi = (__bf16)v;
    chi[(size_t)k * DDIM + d] = hi;
    clo[(size_t)k * DDIM + d] = (__bf16)(v - (float)hi);
    cf32[(size_t)d * KCODES + k] = v;
  }
  cnorm[k] = norm;
}

__global__ __launch_bounds__(256) void prep_x(
    const float* __restrict__ inputs,  // [64][16384]
    __bf16* __restrict__ xhi,          // [16384][64]
    __bf16* __restrict__ xlo,          // [16384][64]
    float*  __restrict__ xnorm)        // [16384]
{
  const int n = blockIdx.x * 256 + threadIdx.x;
  float norm = 0.f;
#pragma unroll
  for (int d = 0; d < DDIM; ++d) {
    float v = inputs[(size_t)d * NROWS + n];
    norm += v * v;
    __bf16 hi = (__bf16)v;
    xhi[(size_t)n * DDIM + d] = hi;
    xlo[(size_t)n * DDIM + d] = (__bf16)(v - (float)hi);
  }
  xnorm[n] = norm;
}

// ---------------------------------------------------------------------------
// vq_argmin: 8 waves/WG, 32 rows/wave, A frags pinned in VGPRs.
// Codebook streamed via TDM into double-buffered LDS (72KB of 320KB).
// One wave drives the DMA; one s_wait_tensorcnt + one barrier per chunk.
// ---------------------------------------------------------------------------
__global__ __launch_bounds__(256) void vq_argmin(
    const __bf16* __restrict__ xhi, const __bf16* __restrict__ xlo,
    const float*  __restrict__ xnorm,
    const __bf16* __restrict__ chi, const __bf16* __restrict__ clo,
    const float*  __restrict__ cnorm,
    float* __restrict__ dist_out,    // [16384]
    float* __restrict__ indf_out,    // [16384] (indices as f32)
    int*   __restrict__ indi_out)    // [16384] (indices as i32)
{
  __shared__ __align__(16) unsigned char lds[2][2 * CHUNK * BPAD];  // [buf][hi|lo]

  const int tid  = threadIdx.x;
  const int wave = tid >> 5;
  const int lane = tid & 31;
  const int h    = lane >> 4;
  const int ln   = lane & 15;
  const int rowBase = blockIdx.x * 256 + wave * 32;   // 2 row-tiles of 16

  // ---- A fragments (pinned): 2 tiles x (hi,lo) x 2 K-halves
  Frag aHi[2][2], aLo[2][2];
#pragma unroll
  for (int t = 0; t < 2; ++t) {
    const int row = rowBase + t * 16 + ln;
#pragma unroll
    for (int kh = 0; kh < 2; ++kh) {
      const unsigned char* p = (const unsigned char*)xhi + (size_t)row * 128 + kh * 64 + h * 16;
      aHi[t][kh].f[0] = *(const f4*)p;
      aHi[t][kh].f[1] = *(const f4*)(p + 32);
      const unsigned char* q = (const unsigned char*)xlo + (size_t)row * 128 + kh * 64 + h * 16;
      aLo[t][kh].f[0] = *(const f4*)q;
      aLo[t][kh].f[1] = *(const f4*)(q + 32);
    }
  }

  float minv[2][8];
  int   mini[2][8];
#pragma unroll
  for (int t = 0; t < 2; ++t)
#pragma unroll
    for (int v = 0; v < 8; ++v) { minv[t][v] = 3.0e38f; mini[t][v] = 0; }

  // ---- prologue: DMA chunk 0 into buffer 0
  if (wave == 0) {
    const unsigned base0 = (unsigned)(size_t)&lds[0][0];
    tdm_load_chunk((const unsigned char*)chi, base0);
    tdm_load_chunk((const unsigned char*)clo, base0 + CHUNK * BPAD);
    __builtin_amdgcn_s_wait_tensorcnt(0);
  }
  __syncthreads();

  for (int kc = 0; kc < KCODES; kc += CHUNK) {
    const int cur = (kc >> 7) & 1;

    // kick DMA for the next chunk into the other buffer (overlaps with compute)
    if (wave == 0 && kc + CHUNK < KCODES) {
      const unsigned baseN = (unsigned)(size_t)&lds[cur ^ 1][0];
      tdm_load_chunk((const unsigned char*)chi + (size_t)(kc + CHUNK) * 128, baseN);
      tdm_load_chunk((const unsigned char*)clo + (size_t)(kc + CHUNK) * 128, baseN + CHUNK * BPAD);
    }

    unsigned char* ldsHi = &lds[cur][0];
    unsigned char* ldsLo = ldsHi + CHUNK * BPAD;

    for (int c0 = 0; c0 < CHUNK; c0 += 16) {
      Frag bH[2], bL[2];
#pragma unroll
      for (int kh = 0; kh < 2; ++kh) {
        const unsigned char* p = ldsHi + (c0 + ln) * BPAD + kh * 64 + h * 16;
        bH[kh].f[0] = *(const f4*)p;  bH[kh].f[1] = *(const f4*)(p + 32);
        const unsigned char* q = ldsLo + (c0 + ln) * BPAD + kh * 64 + h * 16;
        bL[kh].f[0] = *(const f4*)q;  bL[kh].f[1] = *(const f4*)(q + 32);
      }
      const int   gidx = kc + c0 + ln;
      const float cn   = cnorm[gidx];          // 512B/chunk, L2-resident

#pragma unroll
      for (int t = 0; t < 2; ++t) {
        v8f acc = {0.f, 0.f, 0.f, 0.f, 0.f, 0.f, 0.f, 0.f};
        // dot ~= xhi*chi + xhi*clo + xlo*chi  (error ~2^-18; lo*lo dropped)
        acc = wmma_bf16(aHi[t][0].v, bH[0].v, acc);
        acc = wmma_bf16(aHi[t][1].v, bH[1].v, acc);
        acc = wmma_bf16(aHi[t][0].v, bL[0].v, acc);
        acc = wmma_bf16(aHi[t][1].v, bL[1].v, acc);
        acc = wmma_bf16(aLo[t][0].v, bH[0].v, acc);
        acc = wmma_bf16(aLo[t][1].v, bH[1].v, acc);
#pragma unroll
        for (int v = 0; v < 8; ++v) {
          const float dv = cn - 2.0f * acc[v]; // ||e||^2 - 2 x.e  (||x||^2 added later)
          if (dv < minv[t][v]) { minv[t][v] = dv; mini[t][v] = gidx; }
        }
      }
    }

    if (wave == 0) __builtin_amdgcn_s_wait_tensorcnt(0);  // next buffer landed
    __syncthreads();                                      // everyone done with cur
  }

  // ---- butterfly reduce across the 16 lanes sharing each row (first-index ties)
#pragma unroll
  for (int t = 0; t < 2; ++t) {
#pragma unroll
    for (int v = 0; v < 8; ++v) {
      float mv = minv[t][v];
      int   mi = mini[t][v];
#pragma unroll
      for (int off = 8; off >= 1; off >>= 1) {
        const float ov = __shfl_xor(mv, off, 32);
        const int   oi = __shfl_xor(mi, off, 32);
        if (ov < mv || (ov == mv && oi < mi)) { mv = ov; mi = oi; }
      }
      if (ln == 0) {
        const int row = rowBase + t * 16 + h * 8 + v;   // C-layout: m = v + 8h
        dist_out[row] = mv + xnorm[row];
        indf_out[row] = (float)mi;
        indi_out[row] = mi;
      }
    }
  }
}

// ---------------------------------------------------------------------------
__global__ __launch_bounds__(256) void quant_gather(
    const int*   __restrict__ ind,     // [16384]
    const float* __restrict__ cf32,    // [64][8192]
    float*       __restrict__ out)     // [64][16384]
{
  const int i = blockIdx.x * 256 + threadIdx.x;   // i = d*16384 + n
  const int n = i & (NROWS - 1);
  const int d = i >> 14;
  out[i] = cf32[(size_t)d * KCODES + ind[n]];
}

// ---------------------------------------------------------------------------
extern "C" void kernel_launch(void* const* d_in, const int* in_sizes, int n_in,
                              void* d_out, int out_size, void* d_ws, size_t ws_size,
                              hipStream_t stream) {
  (void)in_sizes; (void)n_in; (void)out_size; (void)ws_size;

  const float* inputs = (const float*)d_in[0];   // [1,64,128,128] f32
  const float* embed  = (const float*)d_in[1];   // [64,8193] f32
  const int*   cnt    = (const int*)  d_in[2];   // [8192] i32

  unsigned char* ws = (unsigned char*)d_ws;
  __bf16* chi  = (__bf16*)ws;  ws += (size_t)KCODES * DDIM * 2;
  __bf16* clo  = (__bf16*)ws;  ws += (size_t)KCODES * DDIM * 2;
  __bf16* xhi  = (__bf16*)ws;  ws += (size_t)NROWS  * DDIM * 2;
  __bf16* xlo  = (__bf16*)ws;  ws += (size_t)NROWS  * DDIM * 2;
  float*  cf32 = (float*)ws;   ws += (size_t)DDIM * KCODES * 4;
  float*  cnorm= (float*)ws;   ws += (size_t)KCODES * 4;
  float*  xnorm= (float*)ws;   ws += (size_t)NROWS * 4;
  int*    indi = (int*)ws;     ws += (size_t)NROWS * 4;

  float* quant_out = (float*)d_out;                         // [64*16384]
  float* dist_out  = quant_out + (size_t)DDIM * NROWS;      // [16384]
  float* indf_out  = dist_out + NROWS;                      // [16384]

  hipLaunchKernelGGL(prep_codes, dim3(KCODES / 256), dim3(256), 0, stream,
                     embed, cnt, chi, clo, cf32, cnorm);
  hipLaunchKernelGGL(prep_x, dim3(NROWS / 256), dim3(256), 0, stream,
                     inputs, xhi, xlo, xnorm);
  hipLaunchKernelGGL(vq_argmin, dim3(NROWS / 256), dim3(256), 0, stream,
                     xhi, xlo, xnorm, chi, clo, cnorm, dist_out, indf_out, indi);
  hipLaunchKernelGGL(quant_gather, dim3((DDIM * NROWS) / 256), dim3(256), 0, stream,
                     indi, cf32, quant_out);
}
